// SelfAttention_65541200937348
// MI455X (gfx1250) — matrix operand order repaired and checked by reading, not compiled, not run on hardware
//
#include <hip/hip_runtime.h>
#include <hip/hip_bf16.h>

// Problem constants (match reference)
#define BB 2
#define SS 2048
#define DD 1024
#define HH 16
#define HD 64

typedef __attribute__((ext_vector_type(16))) __bf16 bf16x16;
typedef __attribute__((ext_vector_type(8)))  __bf16 bf16x8;
typedef __attribute__((ext_vector_type(4)))  __bf16 bf16x4;
typedef __attribute__((ext_vector_type(8)))  float  f32x8;
typedef __attribute__((ext_vector_type(4)))  float  f32x4;
typedef __attribute__((ext_vector_type(4)))  int    i32x4;

// ---- CDNA5 async global->LDS copy (ASYNCcnt) with synchronous fallback ----
#if defined(__AMDGCN__) && __has_builtin(__builtin_amdgcn_global_load_async_to_lds_b128)
#define USE_ASYNC_LDS 1
#else
#define USE_ASYNC_LDS 0
#endif

static __device__ __forceinline__ void cp16_g2l(void* lds_dst, const void* gsrc) {
#if USE_ASYNC_LDS
  __builtin_amdgcn_global_load_async_to_lds_b128(
      (__attribute__((address_space(1))) i32x4*)gsrc,
      (__attribute__((address_space(3))) i32x4*)lds_dst, 0, 0);
#else
  *(bf16x8*)lds_dst = *(const bf16x8*)gsrc;
#endif
}

static __device__ __forceinline__ void async_wait0() {
#if USE_ASYNC_LDS
#if __has_builtin(__builtin_amdgcn_s_wait_asynccnt)
  __builtin_amdgcn_s_wait_asynccnt(0);
#else
  asm volatile("s_wait_asynccnt 0x0" ::: "memory");
#endif
#endif
}

static __device__ __forceinline__ f32x8 wmma_bf16(bf16x16 a, bf16x16 b, f32x8 c) {
  // D = A(16x32 bf16) * B(32x16 bf16) + C(16x16 f32)
  return __builtin_amdgcn_wmma_f32_16x16x32_bf16(false, a, false, b, (short)0, c, false, false);
}

// A fragment (16x32 bf16), row-major source, leading dim ldk (elements).
// lanes 0-15: M=lane, K strips {0..7,16..23}; lanes 16-31: strips {8..15,24..31}.
static __device__ __forceinline__ bf16x16 load_frag_a(const __bf16* base, int ldk, int lane) {
  int m = lane & 15;
  int o = (lane < 16) ? 0 : 8;
  const __bf16* row = base + (size_t)m * ldk + o;
  bf16x8 lo = *(const bf16x8*)(row);
  bf16x8 hi = *(const bf16x8*)(row + 16);
  bf16x16 a;
#pragma unroll
  for (int i = 0; i < 8; ++i) { a[i] = lo[i]; a[i + 8] = hi[i]; }
  return a;
}

// B fragment (32x16 bf16) from an [N][K] row-major source (B-transposed storage):
// lane holds column n = lane&15, K strips {0..7,16..23} / {8..15,24..31}.
static __device__ __forceinline__ bf16x16 load_frag_b(const __bf16* base, int ldk, int lane) {
  int n = lane & 15;
  int o = (lane < 16) ? 0 : 8;
  const __bf16* row = base + (size_t)n * ldk + o;
  bf16x8 lo = *(const bf16x8*)(row);
  bf16x8 hi = *(const bf16x8*)(row + 16);
  bf16x16 b;
#pragma unroll
  for (int i = 0; i < 8; ++i) { b[i] = lo[i]; b[i + 8] = hi[i]; }
  return b;
}

// ---------------------------------------------------------------------------
// GEMM: y[row, e] = sum_d x[row, d] * W[e, d]
// Block tile 256(M) x 64(N), K-step 32. 8 waves, each owns M=32 (2 A frags,
// 8 f32 accumulators) so every B fragment read feeds 2 WMMAs.
// OUT_MODE 0: RoPE + bf16 out [b,h,s,hd]   (Q, K)
// OUT_MODE 1: bf16 out [b,h,s,hd]          (V)
// OUT_MODE 2: f32 out [b,s,e] -> d_out     (final projection)
// ---------------------------------------------------------------------------
template <bool IN_BF16, int OUT_MODE>
__global__ __launch_bounds__(256) void gemm_kernel(const void* __restrict__ xin,
                                                   const float* __restrict__ W,
                                                   void* __restrict__ out) {
  __shared__ __bf16 ldsX[256][32];  // 16 KB
  __shared__ __bf16 ldsW[64][32];   // 4 KB

  const int tid = threadIdx.x;
  const int lane = tid & 31;
  const int wave = tid >> 5;
  const int m0 = blockIdx.x * 256;
  const int n0 = blockIdx.y * 64;

  f32x8 acc[2][4] = {};

  for (int k0 = 0; k0 < DD; k0 += 32) {
    // Stage X tile (256x32): 1024 chunks of 8 elems, 4 per thread.
#pragma unroll
    for (int j = 0; j < 4; ++j) {
      int c = tid + 256 * j;
      int r = c >> 2, seg = (c & 3) * 8;
      if (IN_BF16) {
        cp16_g2l(&ldsX[r][seg], (const __bf16*)xin + (size_t)(m0 + r) * DD + k0 + seg);
      } else {
        const float* p = (const float*)xin + (size_t)(m0 + r) * DD + k0 + seg;
        f32x4 a = *(const f32x4*)p;
        f32x4 b = *(const f32x4*)(p + 4);
        bf16x8 o;
#pragma unroll
        for (int i = 0; i < 4; ++i) { o[i] = (__bf16)a[i]; o[i + 4] = (__bf16)b[i]; }
        *(bf16x8*)&ldsX[r][seg] = o;
      }
    }
    // Stage W tile (64n x 32k): W[e][d] row-major IS the [n][k] layout needed.
    {
      int r = tid >> 2, seg = (tid & 3) * 8;
      const float* p = W + (size_t)(n0 + r) * DD + k0 + seg;
      f32x4 a = *(const f32x4*)p;
      f32x4 b = *(const f32x4*)(p + 4);
      bf16x8 o;
#pragma unroll
      for (int i = 0; i < 4; ++i) { o[i] = (__bf16)a[i]; o[i + 4] = (__bf16)b[i]; }
      *(bf16x8*)&ldsW[r][seg] = o;
      if (k0 + 32 < DD) __builtin_prefetch(p + 32, 0, 1);  // global_prefetch_b8
    }
    if (IN_BF16) async_wait0();
    __syncthreads();

    bf16x16 a0 = load_frag_a(&ldsX[wave * 32][0], 32, lane);
    bf16x16 a1 = load_frag_a(&ldsX[wave * 32 + 16][0], 32, lane);
#pragma unroll
    for (int t = 0; t < 4; ++t) {
      bf16x16 b = load_frag_b(&ldsW[t * 16][0], 32, lane);
      acc[0][t] = wmma_bf16(a0, b, acc[0][t]);
      acc[1][t] = wmma_bf16(a1, b, acc[1][t]);
    }
    __syncthreads();
  }

  const int hi = (lane < 16) ? 0 : 8;
  const int h = n0 >> 6;  // 64-wide N tiles are head-aligned

  // hoist RoPE inv_freq (depends only on t, lane)
  float invf[4];
#pragma unroll
  for (int t = 0; t < 4; ++t) {
    int j = (t * 16 + (lane & 15)) & 31;
    invf[t] = __expf(-(float)j * 0.28782313662425575f);  // 10000^(-j/32)
  }

#pragma unroll
  for (int ms = 0; ms < 2; ++ms) {
    const int mrow = m0 + wave * 32 + ms * 16;
    if (OUT_MODE == 2) {
      float* o = (float*)out;
#pragma unroll
      for (int t = 0; t < 4; ++t)
#pragma unroll
        for (int r = 0; r < 8; ++r) {
          int row = mrow + r + hi;
          int col = n0 + t * 16 + (lane & 15);
          o[(size_t)row * DD + col] = acc[ms][t][r];
        }
    } else {
      float vals[4][8];
#pragma unroll
      for (int t = 0; t < 4; ++t)
#pragma unroll
        for (int r = 0; r < 8; ++r) vals[t][r] = acc[ms][t][r];

      if (OUT_MODE == 0) {
        // rotate_half pairs head dim i with i +/- 32 == accumulator tile t +/- 2
#pragma unroll
        for (int t = 0; t < 4; ++t)
#pragma unroll
          for (int r = 0; r < 8; ++r) {
            int row = mrow + r + hi;
            int s = row & (SS - 1);
            float sn, cs;
            __sincosf((float)s * invf[t], &sn, &cs);
            float other = (t < 2) ? acc[ms][t + 2][r] : acc[ms][t - 2][r];
            vals[t][r] = (t < 2) ? (acc[ms][t][r] * cs - other * sn)
                                 : (acc[ms][t][r] * cs + other * sn);
          }
      }
      __bf16* o = (__bf16*)out;
#pragma unroll
      for (int t = 0; t < 4; ++t)
#pragma unroll
        for (int r = 0; r < 8; ++r) {
          int row = mrow + r + hi;
          int b = row >> 11;       // row / S
          int s = row & (SS - 1);  // row % S
          int i = t * 16 + (lane & 15);
          o[(((size_t)(b * HH + h) * SS + s) << 6) + i] = (__bf16)vals[t][r];
        }
    }
  }
}

// ---------------------------------------------------------------------------
// Flash attention: block = 128 query rows of one (b,h); 8 waves x 16 rows.
// K tile async-staged to LDS (shared by all 8 waves), V transposed in LDS,
// online softmax in f32, P re-laid out through per-wave LDS.
// ---------------------------------------------------------------------------
__global__ __launch_bounds__(256) void attn_kernel(const __bf16* __restrict__ Q,
                                                   const __bf16* __restrict__ K,
                                                   const __bf16* __restrict__ V,
                                                   const int* __restrict__ mask,
                                                   __bf16* __restrict__ Aout) {
  __shared__ __bf16 kLds[64][64];      // K tile [key][d], 8 KB
  __shared__ __bf16 vLds[64][64];      // V^T tile [d][key], 8 KB
  __shared__ __bf16 pLds[8][16][64];   // per-wave P tiles, 16 KB

  const int tid = threadIdx.x;
  const int lane = tid & 31;
  const int wave = tid >> 5;
  const int qt = blockIdx.x;
  const int bh = blockIdx.y;
  const int b = bh >> 4;
  const int h = bh & 15;

  const __bf16* Qb = Q + (size_t)bh * SS * HD;
  const __bf16* Kb = K + (size_t)bh * SS * HD;
  const __bf16* Vb = V + (size_t)bh * SS * HD;
  const int q0 = qt * 128 + wave * 16;

  // Q fragments (16 rows x 64 -> two 16x32 A fragments), in registers for good.
  bf16x16 qa[2];
  {
    int m = lane & 15;
    int o = (lane < 16) ? 0 : 8;
#pragma unroll
    for (int kc = 0; kc < 2; ++kc) {
      const __bf16* row = Qb + (size_t)(q0 + m) * HD + kc * 32 + o;
      bf16x8 lo = *(const bf16x8*)(row);
      bf16x8 hi2 = *(const bf16x8*)(row + 16);
      bf16x16 a;
#pragma unroll
      for (int i = 0; i < 8; ++i) { a[i] = lo[i]; a[i + 8] = hi2[i]; }
      qa[kc] = a;
    }
  }

  float mrow[8], lrow[8];
  f32x8 oacc[4] = {};
#pragma unroll
  for (int r = 0; r < 8; ++r) { mrow[r] = -1e30f; lrow[r] = 0.f; }

  const int hi = (lane < 16) ? 0 : 8;

  for (int kt = 0; kt < SS / 64; ++kt) {
    const int kb = kt * 64;

    // Stage K tile (row-major, byte-identical) via async copy: 512 x 16B chunks.
#pragma unroll
    for (int j = 0; j < 2; ++j) {
      int c = tid + 256 * j;
      int r = c >> 3, seg = (c & 7) * 8;
      cp16_g2l(&kLds[r][seg], Kb + (size_t)(kb + r) * HD + seg);
    }
    // Stage V^T (transpose in LDS write): thread -> 4 keys x 4 d, packed b64 stores.
    {
      int kg = (tid & 15) * 4;        // 4 consecutive keys
      int d0 = (tid >> 4) * 4;        // 4 consecutive d
      bf16x4 vr[4];
#pragma unroll
      for (int kk = 0; kk < 4; ++kk)
        vr[kk] = *(const bf16x4*)(Vb + (size_t)(kb + kg + kk) * HD + d0);
#pragma unroll
      for (int j = 0; j < 4; ++j) {
        bf16x4 pk;
#pragma unroll
        for (int kk = 0; kk < 4; ++kk) pk[kk] = vr[kk][j];
        *(bf16x4*)&vLds[d0 + j][kg] = pk;
      }
    }
    if (kt + 1 < SS / 64)
      __builtin_prefetch(Kb + (size_t)(kb + 64 + (tid >> 2)) * HD, 0, 1);
    async_wait0();
    __syncthreads();

    // S = Q K^T from LDS K tile
    f32x8 sacc[4] = {};
#pragma unroll
    for (int t = 0; t < 4; ++t) {
#pragma unroll
      for (int kc = 0; kc < 2; ++kc) {
        bf16x16 bfr = load_frag_b(&kLds[t * 16][kc * 32], 64, lane);
        sacc[t] = wmma_bf16(qa[kc], bfr, sacc[t]);
      }
    }

    // scale + mask
    float sc[4][8];
    float madd[4];
#pragma unroll
    for (int t = 0; t < 4; ++t) {
      int key = kb + t * 16 + (lane & 15);
      madd[t] = mask[b * SS + key] ? 0.f : -1e30f;
    }

    // online softmax (row r lives across the 16-lane half-group)
#pragma unroll
    for (int r = 0; r < 8; ++r) {
      float mx = -1e30f;
#pragma unroll
      for (int t = 0; t < 4; ++t) {
        sc[t][r] = sacc[t][r] * 0.125f + madd[t];
        mx = fmaxf(mx, sc[t][r]);
      }
#pragma unroll
      for (int msk = 1; msk < 16; msk <<= 1) mx = fmaxf(mx, __shfl_xor(mx, msk, 32));
      float newm = fmaxf(mrow[r], mx);
      float f = __expf(mrow[r] - newm);
      float ps = 0.f;
#pragma unroll
      for (int t = 0; t < 4; ++t) {
        float p = __expf(sc[t][r] - newm);
        sc[t][r] = p;
        ps += p;
      }
#pragma unroll
      for (int msk = 1; msk < 16; msk <<= 1) ps += __shfl_xor(ps, msk, 32);
      lrow[r] = lrow[r] * f + ps;
      mrow[r] = newm;
#pragma unroll
      for (int t = 0; t < 4; ++t) oacc[t][r] *= f;
    }

    // P -> per-wave LDS (C layout -> A layout re-distribution)
#pragma unroll
    for (int t = 0; t < 4; ++t)
#pragma unroll
      for (int r = 0; r < 8; ++r)
        pLds[wave][r + hi][t * 16 + (lane & 15)] = (__bf16)sc[t][r];

    // O += P V
#pragma unroll
    for (int kc = 0; kc < 2; ++kc) {
      bf16x16 pa = load_frag_a(&pLds[wave][0][kc * 32], 64, lane);
#pragma unroll
      for (int t = 0; t < 4; ++t) {
        bf16x16 vb = load_frag_b(&vLds[t * 16][kc * 32], 64, lane);
        oacc[t] = wmma_bf16(pa, vb, oacc[t]);
      }
    }
    __syncthreads();
  }

  // normalize, write bf16 in [b, s, D] (head-interleaved) for out-projection
#pragma unroll
  for (int t = 0; t < 4; ++t)
#pragma unroll
    for (int r = 0; r < 8; ++r) {
      int row = q0 + r + hi;
      float ov = oacc[t][r] / lrow[r];
      int d = t * 16 + (lane & 15);
      Aout[((size_t)(b * SS + row)) * DD + h * HD + d] = (__bf16)ov;
    }
}

// ---------------------------------------------------------------------------
extern "C" void kernel_launch(void* const* d_in, const int* in_sizes, int n_in,
                              void* d_out, int out_size, void* d_ws, size_t ws_size,
                              hipStream_t stream) {
  (void)in_sizes; (void)n_in; (void)out_size; (void)ws_size;
  const float* x    = (const float*)d_in[0];
  const int*   mask = (const int*)d_in[1];
  const float* Wq   = (const float*)d_in[2];
  const float* Wk   = (const float*)d_in[3];
  const float* Wv   = (const float*)d_in[4];
  const float* Wo   = (const float*)d_in[5];

  const size_t TOK = (size_t)BB * SS * DD;  // 4M elements
  __bf16* Qw = (__bf16*)d_ws;
  __bf16* Kw = Qw + TOK;
  __bf16* Vw = Kw + TOK;
  __bf16* Aw = Vw + TOK;

  dim3 gGemm(4096 / 256, 1024 / 64);  // (16, 16)
  gemm_kernel<false, 0><<<gGemm, 256, 0, stream>>>((const void*)x, Wq, (void*)Qw);
  gemm_kernel<false, 0><<<gGemm, 256, 0, stream>>>((const void*)x, Wk, (void*)Kw);
  gemm_kernel<false, 1><<<gGemm, 256, 0, stream>>>((const void*)x, Wv, (void*)Vw);

  dim3 gAttn(SS / 128, BB * HH);  // (16, 32)
  attn_kernel<<<gAttn, 256, 0, stream>>>(Qw, Kw, Vw, mask, Aw);

  gemm_kernel<true, 2><<<gGemm, 256, 0, stream>>>((const void*)Aw, Wo, d_out);
}
